// AttentionSentRNN_3667902071426
// MI455X (gfx1250) — matrix operand order repaired and checked by reading, not compiled, not run on hardware
//
#include <hip/hip_runtime.h>
#include <hip/hip_bf16.h>
#include <math.h>

// ---------------------------------------------------------------------------
// Problem constants (from the reference)
// ---------------------------------------------------------------------------
constexpr int S_  = 200;        // sentences (scan length)
constexpr int B_  = 128;        // batch
constexpr int I_  = 1024;       // GRU input width
constexpr int H_  = 512;        // GRU hidden
constexpr int H2_ = 1024;       // 2*H (bi-directional concat)
constexpr int G3_ = 1536;       // 3*H (gate width)
constexpr int C_  = 5;          // classes
constexpr int BH_ = B_ * H_;    // 65536

// Persistent scan kernel geometry
constexpr int JW_    = 32;              // hidden columns per block
constexpr int NBLK_  = (H_ / JW_) * 2;  // 16 j-blocks x 2 dirs = 32 blocks
constexpr int WROWS_ = 3 * JW_;         // 96 staged weight rows per block

#define USE_TDM 1

typedef __attribute__((ext_vector_type(16))) __bf16 v16bf;
typedef __attribute__((ext_vector_type(8)))  float  v8f;
typedef unsigned int u32x4 __attribute__((ext_vector_type(4)));
typedef int          i32x4 __attribute__((ext_vector_type(4)));
typedef int          i32x8 __attribute__((ext_vector_type(8)));

#define DEVFN __device__ __forceinline__

// ---------------------------------------------------------------------------
// WMMA helpers (CDNA5 wave32, D = A(16x32 bf16) x B(32x16 bf16) + C(16x16 f32))
// ---------------------------------------------------------------------------
DEVFN v8f wmma_bf16(v16bf a, v16bf b, v8f c) {
  return __builtin_amdgcn_wmma_f32_16x16x32_bf16(
      /*neg_a=*/false, a, /*neg_b=*/false, b,
      /*c_mod=*/(short)0, c, /*reuse_a=*/false, /*reuse_b=*/false);
}

// A fragment: 16x32 tile of row-major f32 matrix, cvt->bf16.
// Lanes 0-15 (row M=lane) hold K 0..15; lanes 16-31 hold K 16..31.
DEVFN v16bf load_a_bf16(const float* __restrict__ base, int ld, int m0, int k0) {
  const int lane = threadIdx.x & 31;
  const int r    = lane & 15;
  const int half = lane >> 4;
  const float* p = base + (size_t)(m0 + r) * ld + (k0 + half * 16);
  v16bf f;
#pragma unroll
  for (int i = 0; i < 8; ++i) f[i] = (__bf16)p[i];
#pragma unroll
  for (int i = 0; i < 8; ++i) f[8 + i] = (__bf16)p[8 + i];
  return f;
}

// B fragment: 32x16 tile; lane n holds column n of B == row (n0+n) of the
// [N,K] row-major weight matrix, contiguous K. Lanes 0-15: k0..k0+15;
// lanes 16-31: k0+16..k0+31.
DEVFN v16bf load_b_bf16(const float* __restrict__ W, int ld, int n0, int k0) {
  const int lane = threadIdx.x & 31;
  const int n    = lane & 15;
  const int half = lane >> 4;
  const float* p = W + (size_t)(n0 + n) * ld + (k0 + half * 16);
  v16bf f;
#pragma unroll
  for (int i = 0; i < 16; ++i) f[i] = (__bf16)p[i];
  return f;
}

DEVFN float sigmoidf_(float x) { return 1.0f / (1.0f + __expf(-x)); }

// ---------------------------------------------------------------------------
// Kernel 0: transpose W_sent [K,N] -> WsT [N,K]
// ---------------------------------------------------------------------------
__global__ void transpose_ws_kernel(const float* __restrict__ W,
                                    float* __restrict__ WT) {
  int id = blockIdx.x * blockDim.x + threadIdx.x;
  int k = id >> 10, n = id & 1023;
  WT[(size_t)n * 1024 + k] = W[id];
}

// ---------------------------------------------------------------------------
// Kernel 1: input-side gates, both directions (M=25600, K=1024, N=1536 x2)
// ---------------------------------------------------------------------------
__global__ __launch_bounds__(256)
void gemm_gx_kernel(const float* __restrict__ x,
                    const float* __restrict__ Wih_f,
                    const float* __restrict__ Wih_b,
                    const float* __restrict__ bih_f,
                    const float* __restrict__ bih_b,
                    float* __restrict__ Gf,
                    float* __restrict__ Gb) {
  const int m0  = blockIdx.x * 128;
  const int n0  = blockIdx.y * 128;
  const int dir = blockIdx.z;
  const float* W    = dir ? Wih_b : Wih_f;
  const float* bias = dir ? bih_b : bih_f;
  float*       G    = dir ? Gb    : Gf;

  const int wave = threadIdx.x >> 5;
  const int wm = (wave & 3) * 32;
  const int wn = (wave >> 2) * 64;

  v8f acc[2][4] = {};
  for (int k0 = 0; k0 < I_; k0 += 32) {
    if (k0 + 32 < I_) {   // hint the next K-chunk (global_prefetch_b8)
      __builtin_prefetch(x + (size_t)(m0 + wm) * I_ + k0 + 32, 0, 1);
      __builtin_prefetch(W + (size_t)(n0 + wn) * I_ + k0 + 32, 0, 1);
    }
    v16bf a0 = load_a_bf16(x, I_, m0 + wm,      k0);
    v16bf a1 = load_a_bf16(x, I_, m0 + wm + 16, k0);
    v16bf b0 = load_b_bf16(W, I_, n0 + wn,      k0);
    v16bf b1 = load_b_bf16(W, I_, n0 + wn + 16, k0);
    v16bf b2 = load_b_bf16(W, I_, n0 + wn + 32, k0);
    v16bf b3 = load_b_bf16(W, I_, n0 + wn + 48, k0);
    acc[0][0] = wmma_bf16(a0, b0, acc[0][0]);
    acc[0][1] = wmma_bf16(a0, b1, acc[0][1]);
    acc[0][2] = wmma_bf16(a0, b2, acc[0][2]);
    acc[0][3] = wmma_bf16(a0, b3, acc[0][3]);
    acc[1][0] = wmma_bf16(a1, b0, acc[1][0]);
    acc[1][1] = wmma_bf16(a1, b1, acc[1][1]);
    acc[1][2] = wmma_bf16(a1, b2, acc[1][2]);
    acc[1][3] = wmma_bf16(a1, b3, acc[1][3]);
  }

  const int lane = threadIdx.x & 31;
  const int half = lane >> 4, colL = lane & 15;
#pragma unroll
  for (int i = 0; i < 2; ++i) {
#pragma unroll
    for (int j = 0; j < 4; ++j) {
      const int gm = m0 + wm + i * 16 + half * 8;
      const int gn = n0 + wn + j * 16 + colL;
      const float bv = bias[gn];
#pragma unroll
      for (int v = 0; v < 8; ++v)
        G[(size_t)(gm + v) * G3_ + gn] = acc[i][j][v] + bv;
    }
  }
}

// ---------------------------------------------------------------------------
// Barrier-counter init (re-run every kernel_launch: graph-replay safe)
// ---------------------------------------------------------------------------
__global__ void init_ctr_kernel(unsigned* __restrict__ ctr) {
  ctr[threadIdx.x] = 0u;     // 256 entries cover 200 steps
}

// Inter-step device barrier: one counter per step, release/acquire semantics.
DEVFN void grid_sync(unsigned* ctr) {
  __syncthreads();
  if (threadIdx.x == 0) {
    __threadfence();                        // release h writes
    atomicAdd(ctr, 1u);
    while (__hip_atomic_load(ctr, __ATOMIC_ACQUIRE, __HIP_MEMORY_SCOPE_AGENT)
           < (unsigned)NBLK_) {
      __builtin_amdgcn_s_sleep(2);
    }
    __threadfence();                        // acquire other blocks' h
  }
  __syncthreads();
}

// ---------------------------------------------------------------------------
// TDM: stage this block's Whh slice (3 gate planes x JW_ rows x K=512, f32)
// into LDS with one tensor_load_to_lds. 3D tile: x=K(512), y=JW_(32) rows,
// z=3 planes with plane stride 512*512 elements.
// ---------------------------------------------------------------------------
#if USE_TDM
DEVFN void tdm_load_whh_f32(unsigned lds_byte_addr, const float* gptr) {
  const unsigned long long ga = (unsigned long long)(uintptr_t)gptr;
  u32x4 g0 = {
      1u,                                            // count=1 valid descriptor
      lds_byte_addr,                                 // LDS dest
      (unsigned)(ga & 0xFFFFFFFFu),                  // global_addr[31:0]
      (unsigned)((ga >> 32) & 0x1FFFFFFu) | (2u << 30)  // addr[56:32] | type=2
  };
  i32x8 g1 = {
      (int)(2u << 16),                               // data_size = 4B
      (int)((unsigned)(H_ & 0xFFFF) << 16),          // tensor_dim0 = 512 (lo16)
      (int)((unsigned)(H_ & 0xFFFF) << 16),          // dim0 hi=0 | tensor_dim1=512 lo16
      (int)((unsigned)(H_ & 0xFFFF) << 16),          // dim1 hi=0 | tile_dim0 = 512
      (int)((unsigned)JW_ | (3u << 16)),             // tile_dim1=32 | tile_dim2=3
      (int)H_,                                       // tensor_dim0_stride = 512
      0,                                             // stride0 hi | stride1 lo16 (=0)
      (int)((unsigned)(H_ * H_) >> 16)               // tensor_dim1_stride[47:16] = 4
  };
  i32x4 g2 = { 3, 0, 0, 0 };                         // tensor_dim2 = 3 planes
  i32x4 g3 = { 0, 0, 0, 0 };
#if defined(__clang_major__) && (__clang_major__ >= 23)
  i32x8 g4 = { 0, 0, 0, 0, 0, 0, 0, 0 };
  __builtin_amdgcn_tensor_load_to_lds(g0, g1, g2, g3, g4, 0);
#else
  __builtin_amdgcn_tensor_load_to_lds(g0, g1, g2, g3, 0);
#endif
}
#endif

// ---------------------------------------------------------------------------
// Kernel 2: PERSISTENT bidirectional GRU scan.
// Grid (16, 2) = 32 blocks; block = 32 hidden cols for one direction.
// Whh slice staged once in LDS as bf16 (96 rows x 512 K = 96 KB); per step:
// each wave owns one 16-row M-tile (A loaded once per block), 3 gates x
// 2 N-tiles of WMMA over K=512, fused GRU gate math, then inter-step barrier.
// ---------------------------------------------------------------------------
__global__ __launch_bounds__(256)
void gru_scan_kernel(const float* __restrict__ Gf, const float* __restrict__ Gb,
                     const float* __restrict__ h0,         // [2,B,H] initial state
                     float* __restrict__ hbuf,             // [dir][2][B*H] ping-pong
                     const float* __restrict__ Whh_f, const float* __restrict__ Whh_b,
                     const float* __restrict__ bhh_f, const float* __restrict__ bhh_b,
                     float* __restrict__ outbuf, float* __restrict__ state_out,
                     unsigned* __restrict__ barrier_ctr) {
  extern __shared__ char smem_raw[];
  __bf16* Wsm = (__bf16*)smem_raw;                         // [96][512] bf16

  const int dir = blockIdx.y;
  const float* G   = dir ? Gb    : Gf;
  const float* Whh = dir ? Whh_b : Whh_f;
  const float* bhh = dir ? bhh_b : bhh_f;
  const int j0 = blockIdx.x * JW_;

  // ---- stage weights once ----
#if USE_TDM
  float* Wstage = (float*)(smem_raw + WROWS_ * H_ * sizeof(__bf16));
  if (threadIdx.x < 32) {   // wave 0 issues the TDM op and waits on TENSORcnt
    const unsigned lds_off = __builtin_amdgcn_groupstaticsize() +
                             (unsigned)(WROWS_ * H_ * sizeof(__bf16));
    tdm_load_whh_f32(lds_off, Whh + (size_t)j0 * H_);
    __builtin_amdgcn_s_wait_tensorcnt(0);
  }
  __syncthreads();
  for (int idx = threadIdx.x; idx < WROWS_ * H_; idx += blockDim.x)
    Wsm[idx] = (__bf16)Wstage[idx];
  __syncthreads();
#else
  for (int idx = threadIdx.x; idx < WROWS_ * H_; idx += blockDim.x) {
    const int r = idx >> 9, k = idx & (H_ - 1);
    const int g = r / JW_, jj = r % JW_;
    Wsm[idx] = (__bf16)Whh[(size_t)(g * H_ + j0 + jj) * H_ + k];
  }
  __syncthreads();
#endif

  const int wave = threadIdx.x >> 5;
  const int lane = threadIdx.x & 31;
  const int half = lane >> 4, colL = lane & 15;
  const int mrow = wave * 16;                              // this wave's batch rows

  for (int t = 0; t < S_; ++t) {
    const float* hp = (t == 0) ? (h0 + (size_t)dir * BH_)
                               : (hbuf + (size_t)(dir * 2 + ((t - 1) & 1)) * BH_);
    float* hn = hbuf + (size_t)(dir * 2 + (t & 1)) * BH_;
    const int s = dir ? (S_ - 1 - t) : t;                  // sentence consumed

    v8f acc[3][2] = {};                                    // [gate][ntile]
    for (int k0 = 0; k0 < H_; k0 += 32) {
      v16bf a = load_a_bf16(hp, H_, mrow, k0);
#pragma unroll
      for (int g = 0; g < 3; ++g) {
#pragma unroll
        for (int nt = 0; nt < 2; ++nt) {
          const __bf16* p = Wsm + ((size_t)(g * JW_ + nt * 16 + colL) * H_
                                   + k0 + half * 16);
          v16bf bf = *(const v16bf*)p;                     // ds_load from LDS
          acc[g][nt] = wmma_bf16(a, bf, acc[g][nt]);
        }
      }
    }

#pragma unroll
    for (int nt = 0; nt < 2; ++nt) {
      const int j = j0 + nt * 16 + colL;                   // hidden column
      const float br_ = bhh[j], bz_ = bhh[H_ + j], bn_ = bhh[2 * H_ + j];
#pragma unroll
      for (int v = 0; v < 8; ++v) {
        const int b = mrow + half * 8 + v;
        const float* g = G + (size_t)(s * B_ + b) * G3_;
        const float r   = sigmoidf_(g[j]          + acc[0][nt][v] + br_);
        const float z   = sigmoidf_(g[H_ + j]     + acc[1][nt][v] + bz_);
        const float n   = tanhf    (g[2 * H_ + j] + r * (acc[2][nt][v] + bn_));
        const float hpv = hp[(size_t)b * H_ + j];
        const float hv  = (1.0f - z) * n + z * hpv;
        hn[(size_t)b * H_ + j] = hv;
        outbuf[(size_t)(s * B_ + b) * H2_ + dir * H_ + j] = hv;
        if (t == S_ - 1)
          state_out[(size_t)dir * BH_ + (size_t)b * H_ + j] = hv;
      }
    }

    if (t != S_ - 1) grid_sync(barrier_ctr + t);
  }
}

// ---------------------------------------------------------------------------
// Kernel 3: attention scores (one block per s; 128x1024x1024 WMMA GEMM with
// fused tanh/w_proj epilogue; deterministic shuffle+LDS reduction).
// ---------------------------------------------------------------------------
__global__ __launch_bounds__(256)
void attn_scores_kernel(const float* __restrict__ outbuf,
                        const float* __restrict__ WsT,      // [N=1024, K=1024]
                        const float* __restrict__ bias_sent,
                        const float* __restrict__ w_proj,
                        float* __restrict__ scores) {
  __shared__ float s4[4][B_];
  const int s = blockIdx.x;
  const float* A = outbuf + (size_t)s * B_ * H2_;   // [128, 1024]

  const int wave = threadIdx.x >> 5;
  const int lane = threadIdx.x & 31;
  const int half = lane >> 4, colL = lane & 15;
  const int wc = wave & 3;
  const int mg = (wave >> 2) * 64;

  float part[4][8] = {};

  for (int pass = 0; pass < 16; ++pass) {
    const int n0 = (pass * 4 + wc) * 16;
    v8f acc[4] = {};
    for (int k0 = 0; k0 < H2_; k0 += 32) {
      v16bf bf = load_b_bf16(WsT, H2_, n0, k0);
#pragma unroll
      for (int i = 0; i < 4; ++i) {
        v16bf a = load_a_bf16(A, H2_, mg + i * 16, k0);
        acc[i] = wmma_bf16(a, bf, acc[i]);
      }
    }
    const int ncol = n0 + colL;
    const float wp = w_proj[ncol], bs = bias_sent[ncol];
#pragma unroll
    for (int i = 0; i < 4; ++i)
#pragma unroll
      for (int v = 0; v < 8; ++v)
        part[i][v] += tanhf(acc[i][v] + bs) * wp;
  }

  for (int i = threadIdx.x; i < 4 * B_; i += blockDim.x) s4[i >> 7][i & 127] = 0.0f;
  __syncthreads();

#pragma unroll
  for (int i = 0; i < 4; ++i) {
#pragma unroll
    for (int v = 0; v < 8; ++v) {
      float r = part[i][v];
      r += __shfl_xor(r, 1);
      r += __shfl_xor(r, 2);
      r += __shfl_xor(r, 4);
      r += __shfl_xor(r, 8);
      if (colL == 0) s4[wc][mg + i * 16 + half * 8 + v] = r;
    }
  }
  __syncthreads();
  for (int r = threadIdx.x; r < B_; r += blockDim.x)
    scores[(size_t)s * B_ + r] = s4[0][r] + s4[1][r] + s4[2][r] + s4[3][r];
}

// ---------------------------------------------------------------------------
// Kernel 4: softmax over S per batch column; emit attn and attn.T
// ---------------------------------------------------------------------------
__global__ void softmax_attn_kernel(const float* __restrict__ scores,
                                    float* __restrict__ attn,
                                    float* __restrict__ attnT_out) {
  const int b = threadIdx.x;
  float m = -INFINITY;
  for (int s = 0; s < S_; ++s) m = fmaxf(m, scores[s * B_ + b]);
  float sum = 0.0f;
  for (int s = 0; s < S_; ++s) sum += __expf(scores[s * B_ + b] - m);
  const float inv = 1.0f / sum;
  for (int s = 0; s < S_; ++s) {
    const float a = __expf(scores[s * B_ + b] - m) * inv;
    attn[s * B_ + b] = a;
    attnT_out[(size_t)b * S_ + s] = a;
  }
}

// ---------------------------------------------------------------------------
// Kernel 5: ctx[b,h] = sum_s attn[s,b] * out[s,b,h]
// ---------------------------------------------------------------------------
__global__ void ctx_kernel(const float* __restrict__ attn,
                           const float* __restrict__ outbuf,
                           float* __restrict__ ctx) {
  const int id = blockIdx.x * blockDim.x + threadIdx.x;
  const int b = id >> 10, h = id & 1023;
  float acc = 0.0f;
  for (int s = 0; s < S_; ++s)
    acc += attn[s * B_ + b] * outbuf[(size_t)(s * B_ + b) * H2_ + h];
  ctx[id] = acc;
}

// ---------------------------------------------------------------------------
// Kernel 6: logits = ctx @ W_final^T + b_final ; log_softmax
// ---------------------------------------------------------------------------
__global__ void final_kernel(const float* __restrict__ ctx,
                             const float* __restrict__ Wf,
                             const float* __restrict__ bf,
                             float* __restrict__ out) {
  const int b = threadIdx.x;
  float logits[C_];
#pragma unroll
  for (int c = 0; c < C_; ++c) {
    float acc = bf[c];
    const float* w = Wf + (size_t)c * H2_;
    const float* x = ctx + (size_t)b * H2_;
    for (int k = 0; k < H2_; ++k) acc += x[k] * w[k];
    logits[c] = acc;
  }
  float m = logits[0];
#pragma unroll
  for (int c = 1; c < C_; ++c) m = fmaxf(m, logits[c]);
  float sum = 0.0f;
#pragma unroll
  for (int c = 0; c < C_; ++c) sum += __expf(logits[c] - m);
  const float ls = __logf(sum);
#pragma unroll
  for (int c = 0; c < C_; ++c) out[b * C_ + c] = logits[c] - m - ls;
}

// ---------------------------------------------------------------------------
// Host launcher
// ---------------------------------------------------------------------------
extern "C" void kernel_launch(void* const* d_in, const int* in_sizes, int n_in,
                              void* d_out, int out_size, void* d_ws, size_t ws_size,
                              hipStream_t stream) {
  (void)in_sizes; (void)n_in; (void)out_size; (void)ws_size;

  const float* x       = (const float*)d_in[0];
  const float* state0  = (const float*)d_in[1];
  const float* Wih_f   = (const float*)d_in[2];
  const float* Whh_f   = (const float*)d_in[3];
  const float* bih_f   = (const float*)d_in[4];
  const float* bhh_f   = (const float*)d_in[5];
  const float* Wih_b   = (const float*)d_in[6];
  const float* Whh_b   = (const float*)d_in[7];
  const float* bih_b   = (const float*)d_in[8];
  const float* bhh_b   = (const float*)d_in[9];
  const float* W_sent  = (const float*)d_in[10];
  const float* bias_s  = (const float*)d_in[11];
  const float* w_proj  = (const float*)d_in[12];
  const float* W_final = (const float*)d_in[13];
  const float* b_final = (const float*)d_in[14];

  float* out_logp  = (float*)d_out;
  float* out_state = out_logp + B_ * C_;
  float* out_attnT = out_state + 2 * BH_;

  float* ws = (float*)d_ws;
  size_t off = 0;
  float* Gf     = ws + off; off += (size_t)S_ * B_ * G3_;
  float* Gb     = ws + off; off += (size_t)S_ * B_ * G3_;
  float* outbuf = ws + off; off += (size_t)S_ * B_ * H2_;
  float* hbuf   = ws + off; off += 4 * (size_t)BH_;
  float* scores = ws + off; off += (size_t)S_ * B_;
  float* attn   = ws + off; off += (size_t)S_ * B_;
  float* WsT    = ws + off; off += (size_t)H2_ * H2_;
  float* ctx    = ws + off; off += (size_t)B_ * H2_;
  unsigned* bar = (unsigned*)(ws + off); off += 256;

  // 0) transpose W_sent for contiguous B-fragment loads
  transpose_ws_kernel<<<(H2_ * H2_) / 256, 256, 0, stream>>>(W_sent, WsT);

  // 1) input-side gate GEMM, both directions
  gemm_gx_kernel<<<dim3(S_ * B_ / 128, G3_ / 128, 2), 256, 0, stream>>>(
      x, Wih_f, Wih_b, bih_f, bih_b, Gf, Gb);

  // 2) persistent bidirectional GRU scan (single launch, inter-step barrier)
  init_ctr_kernel<<<1, 256, 0, stream>>>(bar);
#if USE_TDM
  const size_t smem_bytes = (size_t)WROWS_ * H_ * (sizeof(__bf16) + sizeof(float));
#else
  const size_t smem_bytes = (size_t)WROWS_ * H_ * sizeof(__bf16);
#endif
  gru_scan_kernel<<<dim3(H_ / JW_, 2), 256, smem_bytes, stream>>>(
      Gf, Gb, state0, hbuf, Whh_f, Whh_b, bhh_f, bhh_b,
      outbuf, out_state, bar);

  // 3) attention scores
  attn_scores_kernel<<<S_, 256, 0, stream>>>(outbuf, WsT, bias_s, w_proj, scores);

  // 4) softmax over S + attn.T output
  softmax_attn_kernel<<<1, B_, 0, stream>>>(scores, attn, out_attnT);

  // 5) context vector
  ctx_kernel<<<(B_ * H2_) / 256, 256, 0, stream>>>(attn, outbuf, ctx);

  // 6) classifier + log_softmax
  final_kernel<<<1, B_, 0, stream>>>(ctx, W_final, b_final, out_logp);
}
